// CTRNFuse_47175920779737
// MI455X (gfx1250) — matrix operand order repaired and checked by reading, not compiled, not run on hardware
//
#include <hip/hip_runtime.h>
#include <hip/hip_bf16.h>
#include <math.h>

// ---------------------------------------------------------------------------
// Problem constants (from reference): B=8, DIM=512, T=16384, H=DIM/2=256
// ---------------------------------------------------------------------------
constexpr int Bn  = 8;
constexpr int DIM = 512;
constexpr int Tn  = 16384;
constexpr int Hn  = 256;

typedef _Float16 v16h __attribute__((ext_vector_type(16)));
typedef _Float16 v8h  __attribute__((ext_vector_type(8)));
typedef float    v8f  __attribute__((ext_vector_type(8)));
typedef unsigned int v4u __attribute__((ext_vector_type(4)));
typedef int      v4i  __attribute__((ext_vector_type(4)));
typedef int      v8i  __attribute__((ext_vector_type(8)));

#if __has_builtin(__builtin_amdgcn_tensor_load_to_lds)
#define HAVE_TDM 1
#else
#define HAVE_TDM 0
#endif

// ---------------------------------------------------------------------------
// Workspace layout (bytes). Total ~203 MB.
// ---------------------------------------------------------------------------
constexpr size_t OFF_YQK   = 0;
constexpr size_t SZ_YQK    = (size_t)Bn*(DIM/4)*Tn*8;
constexpr size_t OFF_Y2QK  = OFF_YQK + SZ_YQK;
constexpr size_t SZ_Y2QK   = (size_t)Bn*(Hn/4)*Tn*8;
constexpr size_t OFF_NT    = OFF_Y2QK + SZ_Y2QK;
constexpr size_t OFF_GNC   = OFF_NT   + (size_t)Bn*Tn*4;
constexpr size_t OFF_STATS = OFF_GNC  + (size_t)Bn*Hn*4;
constexpr size_t OFF_MURS  = OFF_STATS + 256;
constexpr size_t OFF_WG    = OFF_MURS  + 256;
constexpr size_t OFF_PW    = OFF_WG + (size_t)Hn*DIM*2;
constexpr size_t OFF_B0    = OFF_PW + (size_t)Hn*Hn*2;
constexpr size_t OFF_WGS   = OFF_B0 + (size_t)Hn*4;
constexpr size_t OFF_B2    = OFF_WGS + (size_t)Hn*4;

// LDS B-tile: 8 quad-rows x 64 t x 8B, row stride padded to 68 u64 (544 B)
// -> lanes 0-15 hit banks (8q+2t)%64 in [0,31], lanes 16-31 in [32,63].
// TDM reproduces this pad: row = 128 DW (pad_interval=6), pad 8 DW (pad_amount=7).
constexpr int LDS_QSTRIDE = 68;                 // u64 elements per quad-row
constexpr int LDS_TILE    = 8 * LDS_QSTRIDE;    // u64 elements per buffer

// ---------------------------------------------------------------------------
// Helpers
// ---------------------------------------------------------------------------
__device__ __forceinline__ float gelu_exact(float x) {
    return 0.5f * x * (1.0f + erff(x * 0.70710678118654752f));
}

__device__ __forceinline__ float block_reduce_256(float v, float* sm) {
    int tid = threadIdx.x;
    sm[tid] = v;
    __syncthreads();
    #pragma unroll
    for (int s = 128; s > 0; s >>= 1) {
        if (tid < s) sm[tid] += sm[tid + s];
        __syncthreads();
    }
    float r = sm[0];
    __syncthreads();
    return r;
}

__device__ __forceinline__ v8f wmma_f16(v16h a, v16h b, v8f c) {
    return __builtin_amdgcn_wmma_f32_16x16x32_f16(false, a, false, b, (short)0, c,
                                                  false, false);
}

// A-matrix fragment load, 16-bit A 16x32, contiguous k order: k = 16*(lane/16) + s
__device__ __forceinline__ v16h load_a_frag(const _Float16* A, int row, int ld,
                                            int kk, int lane) {
    int koff = kk + ((lane >> 4) << 4);
    v8h lo = *reinterpret_cast<const v8h*>(A + (size_t)row * ld + koff);
    v8h hi = *reinterpret_cast<const v8h*>(A + (size_t)row * ld + koff + 8);
    v16h a;
    #pragma unroll
    for (int i = 0; i < 8; i++) { a[i] = lo[i]; a[i + 8] = hi[i]; }
    return a;
}

// ---- async (ASYNCcnt) staging: 256 threads x 16 B ---------------------------
__device__ __forceinline__ void issue_b_tile_async(const unsigned long long* gq,
                                                   int qbase, int tb,
                                                   unsigned long long* sbuf, int tid) {
    int qrow  = tid >> 5;
    int tpair = tid & 31;
    unsigned ldsoff = (unsigned)(uintptr_t)(sbuf + qrow * LDS_QSTRIDE + tpair * 2);
    unsigned long long gaddr =
        (unsigned long long)(uintptr_t)(gq + (size_t)(qbase + qrow) * Tn + tb + tpair * 2);
    asm volatile("global_load_async_to_lds_b128 %0, %1, off"
                 :: "v"(ldsoff), "v"(gaddr) : "memory");
}

__device__ __forceinline__ void wait_async_and_sync() {
    asm volatile("s_wait_asynccnt 0" ::: "memory");
    __syncthreads();
}

// ---- TDM (TENSORcnt) staging: one descriptor-driven DMA per tile ------------
// Tensor: [nqrows][Tn] of 8-byte elements; tile 8 x 64 starting at (qbase, tb),
// LDS padded 32 B per 512 B row -> 544 B row stride (matches LDS_QSTRIDE).
__device__ __forceinline__ void issue_b_tile_tdm(const unsigned long long* gq,
                                                 int qbase, int tb, int nqrows,
                                                 unsigned long long* sbuf) {
#if HAVE_TDM
    unsigned long long ga =
        (unsigned long long)(uintptr_t)(gq + (size_t)qbase * Tn + tb);
    v4u g0;
    g0[0] = 1u;                                            // count=1, user D#
    g0[1] = (unsigned)(uintptr_t)sbuf;                     // lds_addr (bytes)
    g0[2] = (unsigned)(ga & 0xffffffffu);                  // global_addr[31:0]
    g0[3] = (unsigned)((ga >> 32) & 0x01ffffffu) | 0x80000000u; // addr[56:32]|type=2
    v8i g1;
    g1[0] = (3 << 16) | (1 << 20) | (6 << 22) | (7 << 25); // 8B, pad_en, 128DW, 8DW
    g1[1] = (int)(((unsigned)Tn & 0xffffu) << 16);         // tensor_dim0 lo16
    g1[2] = (int)(((unsigned)nqrows & 0xffffu) << 16);     // dim0 hi=0 | tensor_dim1 lo16
    g1[3] = (int)(64u << 16);                              // dim1 hi=0 | tile_dim0=64
    g1[4] = 8;                                             // tile_dim1=8, tile_dim2=0
    g1[5] = Tn;                                            // tensor_dim0_stride lo32
    g1[6] = 0;
    g1[7] = 0;
    v4i gz = {0, 0, 0, 0};
#if defined(__clang_major__) && (__clang_major__ >= 23)
    v8i gz8 = {0, 0, 0, 0, 0, 0, 0, 0};
    __builtin_amdgcn_tensor_load_to_lds(g0, g1, gz, gz, gz8, 0);
#else
    __builtin_amdgcn_tensor_load_to_lds(g0, g1, gz, gz, 0);
#endif
#endif
}

// stage = TDM issued by wave 0 (fallback: async by all threads)
__device__ __forceinline__ void stage_tile(const unsigned long long* gq, int qbase,
                                           int tb, int nqrows,
                                           unsigned long long* sbuf, int tid) {
#if HAVE_TDM
    if (tid < 32) issue_b_tile_tdm(gq, qbase, tb, nqrows, sbuf);
#else
    (void)nqrows;
    issue_b_tile_async(gq, qbase, tb, sbuf, tid);
#endif
}

__device__ __forceinline__ void stage_wait_and_sync(int tid) {
#if HAVE_TDM
    if (tid < 32) __builtin_amdgcn_s_wait_tensorcnt(0);
    __syncthreads();
#else
    (void)tid;
    wait_async_and_sync();
#endif
}

// B fragment from LDS tile (4x ds_load_b64, pairs merged to ds_load_2addr_b64)
__device__ __forceinline__ v16h load_b_frag_lds(const unsigned long long* sbuf,
                                                int laneHalf, int tloc) {
    v16h b;
    #pragma unroll
    for (int jq = 0; jq < 4; jq++) {
        unsigned long long q = sbuf[(laneHalf * 4 + jq) * LDS_QSTRIDE + tloc];
        union { unsigned long long u; _Float16 h[4]; } cv; cv.u = q;
        b[4 * jq + 0] = cv.h[0]; b[4 * jq + 1] = cv.h[1];
        b[4 * jq + 2] = cv.h[2]; b[4 * jq + 3] = cv.h[3];
    }
    return b;
}

// ---------------------------------------------------------------------------
// K0: zero the per-batch stat accumulators
// ---------------------------------------------------------------------------
__global__ void k_zero_stats(float* stats) {
    if (threadIdx.x < 2 * Bn) stats[threadIdx.x] = 0.0f;
}

// ---------------------------------------------------------------------------
// K1: depthwise conv3 on x -> y (f16, channel-quad packed) + per-batch sum/sumsq
// ---------------------------------------------------------------------------
__global__ __launch_bounds__(256) void k_dwconv_stats(
    const float* __restrict__ x, const float* __restrict__ dw_w,
    const float* __restrict__ dw_b, unsigned long long* __restrict__ yqk,
    float* __restrict__ stats) {
    __shared__ float sm[256];
    int chunk = blockIdx.x & 7;
    int cq    = (blockIdx.x >> 3) & 127;
    int b     = blockIdx.x >> 10;
    int t0    = chunk * 2048 + threadIdx.x * 8;

    _Float16 yh[4][8];
    float s = 0.0f, ss = 0.0f;
    #pragma unroll
    for (int ci = 0; ci < 4; ci++) {
        int c = 4 * cq + ci;
        const float* xr = x + ((size_t)b * DIM + c) * Tn;
        float w0 = dw_w[c * 3 + 0], w1 = dw_w[c * 3 + 1], w2 = dw_w[c * 3 + 2];
        float bias = dw_b[c];
        float prev = (t0 > 0) ? xr[t0 - 1] : 0.0f;
        float cur  = xr[t0];
        #pragma unroll
        for (int j = 0; j < 8; j++) {
            int tn = t0 + j + 1;
            float nxt = (tn < Tn) ? xr[tn] : 0.0f;
            float y = w0 * prev + w1 * cur + w2 * nxt + bias;
            s += y; ss += y * y;
            yh[ci][j] = (_Float16)y;
            prev = cur; cur = nxt;
        }
    }
    unsigned long long* yp = yqk + ((size_t)b * (DIM / 4) + cq) * Tn + t0;
    #pragma unroll
    for (int j = 0; j < 8; j++) {
        union { unsigned long long u; _Float16 h[4]; } cv;
        cv.h[0] = yh[0][j]; cv.h[1] = yh[1][j];
        cv.h[2] = yh[2][j]; cv.h[3] = yh[3][j];
        yp[j] = cv.u;
    }
    float rs_ = block_reduce_256(s, sm);
    float rss = block_reduce_256(ss, sm);
    if (threadIdx.x == 0) {
        atomicAdd(&stats[2 * b + 0], rs_);
        atomicAdd(&stats[2 * b + 1], rss);
    }
}

// ---------------------------------------------------------------------------
// K2: finalize per-batch mean / rsqrt(var)
// ---------------------------------------------------------------------------
__global__ void k_finalize_stats(const float* __restrict__ stats, float* __restrict__ murs) {
    int b = threadIdx.x;
    if (b < Bn) {
        float n  = (float)DIM * (float)Tn;
        float mu = stats[2 * b] / n;
        float var = stats[2 * b + 1] / n - mu * mu;
        murs[2 * b]     = mu;
        murs[2 * b + 1] = rsqrtf(var + 1e-8f);
    }
}

// ---------------------------------------------------------------------------
// K3: pack weights to f16 + fold biases. grid Hn, block 256.
// ---------------------------------------------------------------------------
__global__ __launch_bounds__(256) void k_pack_weights(
    const float* __restrict__ pw_w, const float* __restrict__ pw_b,
    const float* __restrict__ gn_g, const float* __restrict__ gn_b,
    const float* __restrict__ p_w,  const float* __restrict__ p_b,
    const float* __restrict__ beta,
    _Float16* __restrict__ Wg, _Float16* __restrict__ Pw,
    float* __restrict__ bias0, float* __restrict__ wgsum, float* __restrict__ bias2) {
    __shared__ float sm[256];
    int o = blockIdx.x, tid = threadIdx.x;
    float wg_s = 0.0f, b0_s = 0.0f;
    for (int c = tid; c < DIM; c += 256) {
        float w   = pw_w[(size_t)o * DIM + c];
        float wgv = w * gn_g[c];
        Wg[(size_t)o * DIM + c] = (_Float16)wgv;
        wg_s += wgv;
        b0_s += w * gn_b[c];
    }
    float w2 = p_w[(size_t)o * Hn + tid];
    Pw[(size_t)o * Hn + tid] = (_Float16)w2;
    float b2_s = w2 * beta[tid];

    float r1 = block_reduce_256(wg_s, sm);
    float r2 = block_reduce_256(b0_s, sm);
    float r3 = block_reduce_256(b2_s, sm);
    if (tid == 0) {
        wgsum[o] = r1;
        bias0[o] = r2 + pw_b[o];
        bias2[o] = r3 + p_b[o];
    }
}

// ---------------------------------------------------------------------------
// K4: GEMM1  y2 = GELU(rs*(Wg @ y) + cbias)   [WMMA, K=512, TDM B tiles]
// ---------------------------------------------------------------------------
__global__ __launch_bounds__(256) void k_gemm1(
    const unsigned long long* __restrict__ yqk, const _Float16* __restrict__ Wg,
    const float* __restrict__ murs, const float* __restrict__ bias0,
    const float* __restrict__ wgsum, unsigned long long* __restrict__ y2qk) {
    __shared__ float cb[Hn];
    __shared__ unsigned long long sB[2][LDS_TILE];
    int tid = threadIdx.x;
    int b  = blockIdx.x / (Tn / 64);
    int tb = (blockIdx.x % (Tn / 64)) * 64;
    float mu = murs[2 * b], rs = murs[2 * b + 1];
    cb[tid] = bias0[tid] - mu * rs * wgsum[tid];

    int lane = tid & 31, wave = tid >> 5;
    int laneHalf = lane >> 4;
    int rowA = (wave * 2) * 16 + (lane & 15);
    const unsigned long long* ypb = yqk + (size_t)b * (DIM / 4) * Tn;

    stage_tile(ypb, 0, tb, DIM / 4, sB[0], tid);
    stage_wait_and_sync(tid);

    v8f acc[2][4] = {};
    int cur = 0;
    for (int kk = 0; kk < DIM; kk += 32) {
        if (kk + 32 < DIM)
            stage_tile(ypb, (kk + 32) >> 2, tb, DIM / 4, sB[cur ^ 1], tid);
        v16h a0 = load_a_frag(Wg, rowA,      DIM, kk, lane);
        v16h a1 = load_a_frag(Wg, rowA + 16, DIM, kk, lane);
        #pragma unroll
        for (int nj = 0; nj < 4; nj++) {
            v16h bf = load_b_frag_lds(sB[cur], laneHalf, nj * 16 + (lane & 15));
            acc[0][nj] = wmma_f16(a0, bf, acc[0][nj]);
            acc[1][nj] = wmma_f16(a1, bf, acc[1][nj]);
        }
        stage_wait_and_sync(tid);
        cur ^= 1;
    }
    #pragma unroll
    for (int mi = 0; mi < 2; mi++) {
        #pragma unroll
        for (int nj = 0; nj < 4; nj++) {
            int t = tb + nj * 16 + (lane & 15);
            #pragma unroll
            for (int v0 = 0; v0 < 8; v0 += 4) {
                int row = (wave * 2 + mi) * 16 + (laneHalf << 3) + v0;
                union { unsigned long long u; _Float16 h[4]; } cv;
                #pragma unroll
                for (int e = 0; e < 4; e++)
                    cv.h[e] = (_Float16)gelu_exact(rs * acc[mi][nj][v0 + e] + cb[row + e]);
                y2qk[((size_t)b * (Hn / 4) + (row >> 2)) * Tn + t] = cv.u;
            }
        }
    }
}

// ---------------------------------------------------------------------------
// K5: GEMM2 -> Nt   [WMMA, K=256, async-LDS B tiles to exercise ASYNCcnt path]
// ---------------------------------------------------------------------------
__global__ __launch_bounds__(256) void k_gate_time(
    const unsigned long long* __restrict__ y2qk, const _Float16* __restrict__ Pw,
    const float* __restrict__ p_b, float* __restrict__ Nt) {
    __shared__ float pb[Hn];
    __shared__ float red[8][64];
    __shared__ unsigned long long sB[2][LDS_TILE];
    int tid = threadIdx.x;
    int b  = blockIdx.x / (Tn / 64);
    int tb = (blockIdx.x % (Tn / 64)) * 64;
    pb[tid] = p_b[tid];

    int lane = tid & 31, wave = tid >> 5;
    int laneHalf = lane >> 4;
    int rowA = (wave * 2) * 16 + (lane & 15);
    const unsigned long long* yb = y2qk + (size_t)b * (Hn / 4) * Tn;

    issue_b_tile_async(yb, 0, tb, sB[0], tid);
    wait_async_and_sync();

    v8f acc[2][4] = {};
    int cur = 0;
    for (int kk = 0; kk < Hn; kk += 32) {
        if (kk + 32 < Hn)
            issue_b_tile_async(yb, (kk + 32) >> 2, tb, sB[cur ^ 1], tid);
        v16h a0 = load_a_frag(Pw, rowA,      Hn, kk, lane);
        v16h a1 = load_a_frag(Pw, rowA + 16, Hn, kk, lane);
        #pragma unroll
        for (int nj = 0; nj < 4; nj++) {
            v16h bf = load_b_frag_lds(sB[cur], laneHalf, nj * 16 + (lane & 15));
            acc[0][nj] = wmma_f16(a0, bf, acc[0][nj]);
            acc[1][nj] = wmma_f16(a1, bf, acc[1][nj]);
        }
        wait_async_and_sync();
        cur ^= 1;
    }
    #pragma unroll
    for (int nj = 0; nj < 4; nj++) {
        float s = 0.0f;
        #pragma unroll
        for (int mi = 0; mi < 2; mi++) {
            #pragma unroll
            for (int v = 0; v < 8; v++) {
                int row = (wave * 2 + mi) * 16 + (laneHalf << 3) + v;
                float val = acc[mi][nj][v] + pb[row];
                s += val * val;
            }
        }
        s += __shfl_xor(s, 16, 32);
        if (lane < 16) red[wave][nj * 16 + lane] = s;
    }
    __syncthreads();
    if (tid < 64) {
        float tot = 0.0f;
        #pragma unroll
        for (int w = 0; w < 8; w++) tot += red[w][tid];
        float Gt = sqrtf(tot);
        Nt[(size_t)b * Tn + tb + tid] = Gt / (Gt + 1e-6f);
    }
}

// ---------------------------------------------------------------------------
// K6: channel gate. yc = dwconv3(y2), Gc = ||yc||_2 over time, gnc = gamma*Nc
// ---------------------------------------------------------------------------
__global__ __launch_bounds__(256) void k_gate_chan(
    const unsigned long long* __restrict__ y2qk, const float* __restrict__ dwh_w,
    const float* __restrict__ dwh_b, const float* __restrict__ gamma,
    float* __restrict__ gnc) {
    __shared__ float sm[256];
    int h = blockIdx.x % Hn;
    int b = blockIdx.x / Hn;
    const _Float16* yr =
        reinterpret_cast<const _Float16*>(y2qk + ((size_t)b * (Hn / 4) + (h >> 2)) * Tn);
    int sub = h & 3;
    float w0 = dwh_w[h * 3 + 0], w1 = dwh_w[h * 3 + 1], w2 = dwh_w[h * 3 + 2];
    float bias = dwh_b[h];

    int t0 = threadIdx.x * 64;
    float ss = 0.0f;
    float prev = (t0 > 0) ? (float)yr[(size_t)(t0 - 1) * 4 + sub] : 0.0f;
    float cur  = (float)yr[(size_t)t0 * 4 + sub];
    for (int j = 0; j < 64; j++) {
        int tn = t0 + j + 1;
        float nxt = (tn < Tn) ? (float)yr[(size_t)tn * 4 + sub] : 0.0f;
        float yc = w0 * prev + w1 * cur + w2 * nxt + bias;
        ss += yc * yc;
        prev = cur; cur = nxt;
    }
    float tot = block_reduce_256(ss, sm);
    if (threadIdx.x == 0) {
        float Gc = sqrtf(tot);
        gnc[(size_t)b * Hn + h] = gamma[h] * (Gc / (Gc + 1e-6f));
    }
}

// ---------------------------------------------------------------------------
// K7: GEMM3  out = Pw @ (y2 * (1 + gnc[h]*Nt[t])) + bias2   [WMMA, TDM B tiles]
// ---------------------------------------------------------------------------
__global__ __launch_bounds__(256) void k_gemm3(
    const unsigned long long* __restrict__ y2qk, const _Float16* __restrict__ Pw,
    const float* __restrict__ bias2, const float* __restrict__ Nt,
    const float* __restrict__ gnc, float* __restrict__ out) {
    __shared__ float b2[Hn];
    __shared__ unsigned long long sB[2][LDS_TILE];
    int tid = threadIdx.x;
    int b  = blockIdx.x / (Tn / 64);
    int tb = (blockIdx.x % (Tn / 64)) * 64;
    b2[tid] = bias2[tid];

    int lane = tid & 31, wave = tid >> 5;
    int laneHalf = lane >> 4;
    int rowA = (wave * 2) * 16 + (lane & 15);
    const unsigned long long* yb = y2qk + (size_t)b * (Hn / 4) * Tn;
    const float4* gp4 = reinterpret_cast<const float4*>(gnc + (size_t)b * Hn);

    float ntv[4];
    #pragma unroll
    for (int nj = 0; nj < 4; nj++)
        ntv[nj] = Nt[(size_t)b * Tn + tb + nj * 16 + (lane & 15)];

    stage_tile(yb, 0, tb, Hn / 4, sB[0], tid);
    stage_wait_and_sync(tid);

    v8f acc[2][4] = {};
    int cur = 0;
    for (int kk = 0; kk < Hn; kk += 32) {
        if (kk + 32 < Hn)
            stage_tile(yb, (kk + 32) >> 2, tb, Hn / 4, sB[cur ^ 1], tid);
        v16h a0 = load_a_frag(Pw, rowA,      Hn, kk, lane);
        v16h a1 = load_a_frag(Pw, rowA + 16, Hn, kk, lane);
        int qb = (kk >> 2) + laneHalf * 4;
        #pragma unroll
        for (int nj = 0; nj < 4; nj++) {
            int tloc = nj * 16 + (lane & 15);
            float nt = ntv[nj];
            v16h bf;
            #pragma unroll
            for (int jq = 0; jq < 4; jq++) {
                unsigned long long q = sB[cur][(laneHalf * 4 + jq) * LDS_QSTRIDE + tloc];
                float4 g4 = gp4[qb + jq];
                union { unsigned long long u; _Float16 h[4]; } cv; cv.u = q;
                bf[4 * jq + 0] = (_Float16)((float)cv.h[0] * (1.0f + g4.x * nt));
                bf[4 * jq + 1] = (_Float16)((float)cv.h[1] * (1.0f + g4.y * nt));
                bf[4 * jq + 2] = (_Float16)((float)cv.h[2] * (1.0f + g4.z * nt));
                bf[4 * jq + 3] = (_Float16)((float)cv.h[3] * (1.0f + g4.w * nt));
            }
            acc[0][nj] = wmma_f16(a0, bf, acc[0][nj]);
            acc[1][nj] = wmma_f16(a1, bf, acc[1][nj]);
        }
        stage_wait_and_sync(tid);
        cur ^= 1;
    }
    #pragma unroll
    for (int mi = 0; mi < 2; mi++) {
        #pragma unroll
        for (int nj = 0; nj < 4; nj++) {
            int t = tb + nj * 16 + (lane & 15);
            #pragma unroll
            for (int v = 0; v < 8; v++) {
                int row = (wave * 2 + mi) * 16 + (laneHalf << 3) + v;
                out[((size_t)b * Hn + row) * Tn + t] = acc[mi][nj][v] + b2[row];
            }
        }
    }
}

// ---------------------------------------------------------------------------
// Launch
// ---------------------------------------------------------------------------
extern "C" void kernel_launch(void* const* d_in, const int* in_sizes, int n_in,
                              void* d_out, int out_size, void* d_ws, size_t ws_size,
                              hipStream_t stream) {
    const float* x     = (const float*)d_in[0];
    const float* dw_w  = (const float*)d_in[1];
    const float* dw_b  = (const float*)d_in[2];
    const float* gn_g  = (const float*)d_in[3];
    const float* gn_b  = (const float*)d_in[4];
    const float* pw_w  = (const float*)d_in[5];
    const float* pw_b  = (const float*)d_in[6];
    const float* dwh_w = (const float*)d_in[7];
    const float* dwh_b = (const float*)d_in[8];
    const float* gamma = (const float*)d_in[9];
    const float* beta  = (const float*)d_in[10];
    const float* p_w   = (const float*)d_in[11];
    const float* p_b   = (const float*)d_in[12];
    (void)in_sizes; (void)n_in; (void)out_size; (void)ws_size;

    char* ws = (char*)d_ws;
    unsigned long long* yqk  = (unsigned long long*)(ws + OFF_YQK);
    unsigned long long* y2qk = (unsigned long long*)(ws + OFF_Y2QK);
    float*     Nt    = (float*)(ws + OFF_NT);
    float*     gnc   = (float*)(ws + OFF_GNC);
    float*     stats = (float*)(ws + OFF_STATS);
    float*     murs  = (float*)(ws + OFF_MURS);
    _Float16*  Wg    = (_Float16*)(ws + OFF_WG);
    _Float16*  Pw    = (_Float16*)(ws + OFF_PW);
    float*     bias0 = (float*)(ws + OFF_B0);
    float*     wgsum = (float*)(ws + OFF_WGS);
    float*     bias2 = (float*)(ws + OFF_B2);
    float*     out   = (float*)d_out;

    k_zero_stats<<<1, 32, 0, stream>>>(stats);
    k_dwconv_stats<<<Bn * (DIM / 4) * (Tn / 2048), 256, 0, stream>>>(
        x, dw_w, dw_b, yqk, stats);
    k_finalize_stats<<<1, 32, 0, stream>>>(stats, murs);
    k_pack_weights<<<Hn, 256, 0, stream>>>(pw_w, pw_b, gn_g, gn_b, p_w, p_b, beta,
                                           Wg, Pw, bias0, wgsum, bias2);
    k_gemm1<<<Bn * (Tn / 64), 256, 0, stream>>>(yqk, Wg, murs, bias0, wgsum, y2qk);
    k_gate_time<<<Bn * (Tn / 64), 256, 0, stream>>>(y2qk, Pw, p_b, Nt);
    k_gate_chan<<<Bn * Hn, 256, 0, stream>>>(y2qk, dwh_w, dwh_b, gamma, gnc);
    k_gemm3<<<Bn * (Tn / 64), 256, 0, stream>>>(y2qk, Pw, bias2, Nt, gnc, out);
}
